// RelationMessagePassing_56788057588327
// MI455X (gfx1250) — matrix operand order repaired and checked by reading, not compiled, not run on hardware
//
#include <hip/hip_runtime.h>
#include <math.h>

typedef _Float16 half_t;
typedef __attribute__((ext_vector_type(16))) _Float16 v16h;
typedef __attribute__((ext_vector_type(4)))  _Float16 h4;
typedef __attribute__((ext_vector_type(8)))  float    v8f;

#define DEV __device__ __forceinline__

// ---------- gfx1250 async global->LDS copy (ASYNCcnt path), guarded ----------
#if __has_builtin(__builtin_amdgcn_global_load_async_to_lds_b64)
#define HAVE_ASYNC 1
typedef int v2i __attribute__((vector_size(8)));
typedef __attribute__((address_space(1))) v2i* g_v2i_p;   // global
typedef __attribute__((address_space(3))) v2i* l_v2i_p;   // LDS
DEV void async_cp8(const half_t* g, half_t* l){
  __builtin_amdgcn_global_load_async_to_lds_b64((g_v2i_p)(void*)g, (l_v2i_p)(void*)l, 0, 0);
}
#else
#define HAVE_ASYNC 0
DEV void async_cp8(const half_t* g, half_t* l){ *(uint2*)l = *(const uint2*)g; }
#endif

DEV void wait_async(){
#if HAVE_ASYNC
#if __has_builtin(__builtin_amdgcn_s_wait_asynccnt)
  __builtin_amdgcn_s_wait_asynccnt(0);
#else
  asm volatile("s_wait_asynccnt 0" ::: "memory");
#endif
#endif
}

// ---- monotonic uint encoding of float for atomicMax on signed floats ----
DEV unsigned encf(float f){ unsigned u=__float_as_uint(f); return (u&0x80000000u)? ~u : (u|0x80000000u); }
DEV float    decf(unsigned e){ return __uint_as_float((e&0x80000000u)? (e&0x7FFFFFFFu) : ~e); }

DEV float mishf(float x){ float sp = (x>20.f)? x : log1pf(expf(x)); return x*tanhf(sp); }

// ---- A fragment: 16x32 f16, ISA 7.12.2 layout ----
DEV v16h ld_a(const half_t* Xs, int xs, int rowBase, int lane, int k0){
  const half_t* p = Xs + (size_t)(rowBase + (lane&15))*xs + k0 + ((lane>>4)<<3);
  union { v16h v; uint4 q[2]; } u;
  u.q[0] = *(const uint4*)p;
  u.q[1] = *(const uint4*)(p+16);
  return u.v;
}
// ---- B fragment: 32x16 f16; Ws[n][kk] chunk-transposed, stride 36 halves ----
DEV v16h ld_b(const half_t* Ws, int nt, int lane){
  const half_t* p = Ws + (size_t)(nt*16 + (lane&15))*36 + ((lane>>4)<<3);
  union { v16h v; uint2 q[4]; } u;
  u.q[0]=*(const uint2*)p;      u.q[1]=*(const uint2*)(p+4);
  u.q[2]=*(const uint2*)(p+16);  u.q[3]=*(const uint2*)(p+20);
  return u.v;
}

// K-loop GEMM: acc[NP/16] += Xs(16 rows/wave, K=D) @ Wt (f16, pre-transposed: Wt[n*D+k])
template<int D, int NP>
DEV void gemm_panel(const half_t* Xs, half_t* Ws, const half_t* __restrict__ Wt,
                    int rowBase, int lane, int tid, v8f* acc){
  constexpr int NT = NP/16;
  for (int k0 = 0; k0 < D; k0 += 32){
    __syncthreads();                       // Ws reuse fence
    for (int g = tid; g < NP*8; g += 256){ // copy Wt[:, k0:k0+32] chunk, 4 halves/item
      int n = g % NP, c = (g / NP)*4;
      async_cp8(Wt + (size_t)n*D + k0 + c, Ws + (size_t)n*36 + c);
    }
    wait_async();
    __syncthreads();
    v16h a = ld_a(Xs, D+8, rowBase, lane, k0);
    #pragma unroll
    for (int nt = 0; nt < NT; ++nt){
      v16h b = ld_b(Ws, nt, lane);
      acc[nt] = __builtin_amdgcn_wmma_f32_16x16x32_f16(false, a, false, b,
                                                       (short)0, acc[nt], false, false);
    }
  }
}

// MODE: 0 = scatter-max into m_u32, 1 = scatter-add exp(12*(v-m)) into s, 2 = store final out
// ARITY: 1/2/3 gather mode; 0 = update mode (concat [max_msg, obj_emb])
template<int D, int DOUT, int ARITY, int MODE>
__global__ __launch_bounds__(256)
void mlp_kernel(const half_t* __restrict__ obj16, const int* __restrict__ idx, int T,
                const half_t* __restrict__ rwt, const float* __restrict__ rb,
                const half_t* __restrict__ owt, const float* __restrict__ ob,
                unsigned int* __restrict__ m_u32, float* __restrict__ s_buf,
                float* __restrict__ out)
{
  constexpr int XS  = D + 8;      // halves per Xs row (16B-aligned rows, bank-clean)
  constexpr int NT1 = D/16;
  constexpr int NT2 = DOUT/16;
  __shared__ __align__(16) half_t Xs[128*XS];
  __shared__ __align__(16) half_t Ws[D*36];

  const int tid  = threadIdx.x;
  const int lane = tid & 31;
  const int wave = tid >> 5;
  const int row0 = blockIdx.x * 128;
  const int wrb  = wave * 16;     // wave's local M-strip base

  // ---- stage X tile (f16 copies; async where possible) ----
  constexpr int C4 = D/4;
  for (int g = tid; g < 128*C4; g += 256){
    int r = g / C4, col = (g % C4)*4;
    int t = row0 + r;
    half_t* dst = Xs + (size_t)r*XS + col;
    if constexpr (ARITY > 0){
      if (t < T){
        int o = idx[(size_t)t*ARITY + (col>>6)];
        async_cp8(obj16 + (size_t)o*64 + (col & 63), dst);
      } else {
        *(uint2*)dst = make_uint2(0u, 0u);          // zero pad rows (discarded later)
      }
    } else {
      if (col < 64){           // max_msg = log(s + 1e-16)/12 + m
        size_t base = (size_t)t*64 + col;
        const float inv12 = 1.f/12.f;
        float a = logf(s_buf[base+0] + 1e-16f)*inv12 + decf(m_u32[base+0]);
        float b = logf(s_buf[base+1] + 1e-16f)*inv12 + decf(m_u32[base+1]);
        float c = logf(s_buf[base+2] + 1e-16f)*inv12 + decf(m_u32[base+2]);
        float d = logf(s_buf[base+3] + 1e-16f)*inv12 + decf(m_u32[base+3]);
        h4 hv = { (half_t)a, (half_t)b, (half_t)c, (half_t)d };
        *(h4*)dst = hv;
      } else {
        async_cp8(obj16 + (size_t)t*64 + (col - 64), dst);
      }
    }
  }
  wait_async();   // all Xs async writes landed before the barrier inside gemm_panel

  // ---- GEMM1: h = x @ rw ----
  v8f acc[NT1];
  v8f zero = {};
  #pragma unroll
  for (int nt = 0; nt < NT1; ++nt) acc[nt] = zero;
  gemm_panel<D, D>(Xs, Ws, rwt, wrb, lane, tid, acc);

  // ---- epilogue1: y = x + mish(h + rb); overwrite Xs (wave-private rows) ----
  {
    const int jrow = (lane>>4)<<3;
    #pragma unroll
    for (int nt = 0; nt < NT1; ++nt){
      int n = nt*16 + (lane&15);
      float bn = rb[n];
      #pragma unroll
      for (int j = 0; j < 8; ++j){
        int r = wrb + j + jrow;
        float x = (float)Xs[(size_t)r*XS + n];
        Xs[(size_t)r*XS + n] = (half_t)(x + mishf(acc[nt][j] + bn));
      }
    }
  }

  // ---- GEMM2: z = y @ ow ----
  #pragma unroll
  for (int nt = 0; nt < NT2; ++nt) acc[nt] = zero;
  gemm_panel<D, DOUT>(Xs, Ws, owt, wrb, lane, tid, acc);

  // ---- epilogue2: + ob, then scatter-max / scatter-sum / store ----
  {
    const int jrow = (lane>>4)<<3;
    #pragma unroll
    for (int nt = 0; nt < NT2; ++nt){
      int c = nt*16 + (lane&15);
      float bn = ob[c];
      #pragma unroll
      for (int j = 0; j < 8; ++j){
        int tr = row0 + wrb + j + jrow;
        if (tr >= T) continue;
        float v = acc[nt][j] + bn;
        if constexpr (MODE == 2){
          out[(size_t)tr*DOUT + c] = v;
        } else {
          int o = idx[(size_t)tr*ARITY + (c>>6)];
          size_t pos = (size_t)o*64 + (c & 63);
          if constexpr (MODE == 0){
            atomicMax(&m_u32[pos], encf(v));
          } else {
            float m = decf(m_u32[pos]);
            atomicAdd(&s_buf[pos], __expf(12.f*(v - m)));   // v <= m, no overflow
          }
        }
      }
    }
  }
}

// ---- prep / init kernels ----
__global__ void k_prep_emb(const float* __restrict__ src, half_t* __restrict__ dst, int n4){
  int i = blockIdx.x*256 + threadIdx.x;
  if (i < n4){
    float4 v = *(const float4*)(src + (size_t)i*4);
    h4 hv = { (half_t)v.x, (half_t)v.y, (half_t)v.z, (half_t)v.w };
    *(h4*)(dst + (size_t)i*4) = hv;
  }
}
// transpose+convert: Wt[n*K + k] = (f16) W[k*NP + n]   (W is fi=K x fo=NP, row-major)
__global__ void k_prep_w(const float* __restrict__ W, half_t* __restrict__ Wt, int K, int NP){
  int j = blockIdx.x*256 + threadIdx.x;
  if (j < K*NP){
    int n = j / K, k = j % K;
    Wt[j] = (half_t)W[(size_t)k*NP + n];
  }
}
__global__ void k_zero_flags(unsigned* flags, int n){
  int i = blockIdx.x*256 + threadIdx.x; if (i < n) flags[i] = 0u;
}
__global__ void k_set_flags(const int* __restrict__ idx, unsigned* flags, int L){
  int i = blockIdx.x*256 + threadIdx.x; if (i < L) flags[idx[i]] = 1u;
}
__global__ void k_init_ms(const unsigned* __restrict__ flags, unsigned* m, float* s, int total){
  int i = blockIdx.x*256 + threadIdx.x;
  if (i < total){
    m[i] = flags[i>>6] ? encf(-INFINITY) : encf(0.f);   // written0 -> -inf, else 0
    s[i] = 0.f;                                         // 1e-16 added at read time
  }
}

extern "C" void kernel_launch(void* const* d_in, const int* in_sizes, int n_in,
                              void* d_out, int out_size, void* d_ws, size_t ws_size,
                              hipStream_t stream) {
  (void)n_in; (void)out_size; (void)ws_size;
  const float* obj = (const float*)d_in[0];
  const int* i0 = (const int*)d_in[1];
  const int* i1 = (const int*)d_in[2];
  const int* i2 = (const int*)d_in[3];
  const float* w[16];
  for (int k = 0; k < 16; ++k) w[k] = (const float*)d_in[4+k];
  // w[0..3]=r0 {res_w,res_b,out_w,out_b}; w[4..7]=r1; w[8..11]=r2; w[12..15]=upd

  const int N  = in_sizes[0] / 64;
  const int L0 = in_sizes[1];
  const int T0 = L0, T1 = in_sizes[2]/2, T2 = in_sizes[3]/3;
  const int tot = N*64;

  // workspace layout
  char* p = (char*)d_ws;
  unsigned* m_u  = (unsigned*)p;            p += (size_t)tot*4;
  float*    s    = (float*)p;               p += (size_t)tot*4;
  unsigned* flg  = (unsigned*)p;            p += (size_t)N*4;
  half_t*   o16  = (half_t*)p;              p += (size_t)tot*2;
  // f16 transposed weights: {rw0,ow0,rw1,ow1,rw2,ow2,rwu,owu}
  const int wK [8] = {64,64,128,128,192,192,128,128};   // fi (= D)
  const int wNP[8] = {64,64,128,128,192,192,128, 64};   // fo (= panel N)
  const int wsrc[8] = {0,2,4,6,8,10,12,14};
  half_t* wt[8];
  for (int k = 0; k < 8; ++k){ wt[k] = (half_t*)p; p += (size_t)wK[k]*wNP[k]*2; }
  float* out = (float*)d_out;

  // ---- prep ----
  k_prep_emb<<<(tot/4+255)/256, 256, 0, stream>>>(obj, o16, tot/4);
  for (int k = 0; k < 8; ++k){
    int e = wK[k]*wNP[k];
    k_prep_w<<<(e+255)/256, 256, 0, stream>>>(w[wsrc[k]], wt[k], wK[k], wNP[k]);
  }
  k_zero_flags<<<(N+255)/256, 256, 0, stream>>>(flg, N);
  k_set_flags <<<(L0+255)/256, 256, 0, stream>>>(i0, flg, L0);
  k_init_ms   <<<(tot+255)/256, 256, 0, stream>>>(flg, m_u, s, tot);

  // pass A: scatter-max of messages into m (recompute, no message buffer)
  mlp_kernel< 64, 64,1,0><<<(T0+127)/128,256,0,stream>>>(o16,i0,T0,wt[0],w[1],wt[1],w[3],  m_u,s,nullptr);
  mlp_kernel<128,128,2,0><<<(T1+127)/128,256,0,stream>>>(o16,i1,T1,wt[2],w[5],wt[3],w[7],  m_u,s,nullptr);
  mlp_kernel<192,192,3,0><<<(T2+127)/128,256,0,stream>>>(o16,i2,T2,wt[4],w[9],wt[5],w[11], m_u,s,nullptr);
  // pass B: scatter-add exp(12*(msg - m))
  mlp_kernel< 64, 64,1,1><<<(T0+127)/128,256,0,stream>>>(o16,i0,T0,wt[0],w[1],wt[1],w[3],  m_u,s,nullptr);
  mlp_kernel<128,128,2,1><<<(T1+127)/128,256,0,stream>>>(o16,i1,T1,wt[2],w[5],wt[3],w[7],  m_u,s,nullptr);
  mlp_kernel<192,192,3,1><<<(T2+127)/128,256,0,stream>>>(o16,i2,T2,wt[4],w[9],wt[5],w[11], m_u,s,nullptr);
  // update MLP on [max_msg, obj_emb] -> output
  mlp_kernel<128, 64,0,2><<<(N+127)/128,256,0,stream>>>(o16,nullptr,N,wt[6],w[13],wt[7],w[15],m_u,s,out);
}
